// MambaBlock_88304527606275
// MI455X (gfx1250) — matrix operand order repaired and checked by reading, not compiled, not run on hardware
//
#include <hip/hip_runtime.h>
#include <hip/hip_bf16.h>

typedef unsigned short     u16;
typedef unsigned int       u32;
typedef unsigned long long u64;

typedef __attribute__((ext_vector_type(16))) __bf16 v16bf;
typedef __attribute__((ext_vector_type(8)))  float  v8f;

// ---------------- problem constants ----------------
static constexpr int D_MODEL   = 1024;
static constexpr int D_INNER   = 2048;
static constexpr int HEADDIM   = 64;
static constexpr int NHEADS    = 32;
static constexpr int D_STATE   = 128;
static constexpr int CONV_DIM  = 2304;   // D_INNER + 2*D_STATE
static constexpr int D_IN_PROJ = 4384;   // 2*D_INNER + 2*D_STATE + NHEADS
static constexpr int NP_IN     = 4480;   // D_IN_PROJ padded to multiple of 128
static constexpr int BATCH     = 4;
static constexpr int SEQLEN    = 2048;
static constexpr int T_TOK     = BATCH * SEQLEN; // 8192
static constexpr float EPS     = 1e-5f;

// ---------------- small helpers ----------------
__device__ __forceinline__ u16 f32_bf16(float f) {
  u32 u = __float_as_uint(f);
  u += 0x7FFFu + ((u >> 16) & 1u);   // round-to-nearest-even
  return (u16)(u >> 16);
}
__device__ __forceinline__ float sigmoidf_(float x) { return 1.f / (1.f + __expf(-x)); }
__device__ __forceinline__ float siluf_(float x)    { return x * sigmoidf_(x); }

// --------- CDNA5 async global->LDS copies (ASYNCcnt-tracked) ---------
// Low 32 bits of a flat shared pointer are the LDS byte address (ISA 10.2).
__device__ __forceinline__ u32 lds_addr(const void* p) { return (u32)(size_t)p; }

__device__ __forceinline__ void async_b128(u32 lds, const void* g) {
  asm volatile("global_load_async_to_lds_b128 %0, %1, off"
               :: "v"(lds), "v"((u64)(size_t)g) : "memory");
}
__device__ __forceinline__ void async_b32(u32 lds, const void* g) {
  asm volatile("global_load_async_to_lds_b32 %0, %1, off"
               :: "v"(lds), "v"((u64)(size_t)g) : "memory");
}
#if defined(__has_builtin) && __has_builtin(__builtin_amdgcn_s_wait_asynccnt)
#define WAIT_ASYNC(n) __builtin_amdgcn_s_wait_asynccnt(n)
#else
#define WAIT_ASYNC(n) asm volatile("s_wait_asynccnt %0" :: "i"(n) : "memory")
#endif

// ---------------- fp32 -> bf16 convert ----------------
__global__ void cvt_bf16_kernel(const float* __restrict__ src, u16* __restrict__ dst, size_t n) {
  size_t i = (size_t)blockIdx.x * 256 + threadIdx.x;
  if (i < n) dst[i] = f32_bf16(src[i]);
}

// ---------------- transpose K x N fp32  ->  NP x K bf16 (zero padded rows) ----------------
__global__ __launch_bounds__(256) void transpose_bf16_kernel(
    const float* __restrict__ W, u16* __restrict__ Wt, int K, int N) {
  __shared__ float tile[32][33];
  const int tx = threadIdx.x & 31, ty = threadIdx.x >> 5;
  const int nb = blockIdx.x, kb = blockIdx.y;
#pragma unroll
  for (int r = 0; r < 4; ++r) {
    int k = kb * 32 + ty + r * 8;
    int n = nb * 32 + tx;
    tile[ty + r * 8][tx] = (n < N) ? W[(size_t)k * N + n] : 0.f;
  }
  __syncthreads();
#pragma unroll
  for (int r = 0; r < 4; ++r) {
    int n = nb * 32 + ty + r * 8;
    int k = kb * 32 + tx;
    Wt[(size_t)n * K + k] = f32_bf16(tile[tx][ty + r * 8]);
  }
}

// ---------------- bf16 WMMA GEMM: C[M x N] = A[M x K] * Bt[N x K]^T ----------------
// 128x128 workgroup tile, 8 waves of 32x64, K-step 32.
// Double-buffered LDS tiles filled by global_load_async_to_lds_b128; the async
// engine fetches tile k+1 while the wave runs the 8 WMMAs of tile k.
// EPI: 0 = store fp32 C; 1 = C = acc + bias[col]; 2 = r = acc + resid, store fp32 + bf16
static constexpr int LDSH = 40;   // padded half-stride (80B rows, 16B aligned)

template <int EPI>
__global__ __launch_bounds__(256) void gemm_bf16_wmma(
    const u16* __restrict__ A, const u16* __restrict__ Bt, float* __restrict__ C,
    int N, int K,
    const float* __restrict__ bias, const float* __restrict__ resid, u16* __restrict__ outh) {
  __shared__ u16 lA[2][128 * LDSH];
  __shared__ u16 lB[2][128 * LDSH];

  const int tid     = threadIdx.x;
  const int rowBase = blockIdx.y * 128;
  const int colBase = blockIdx.x * 128;
  const int wave  = tid >> 5;
  const int lane  = tid & 31;
  const int waveM = wave >> 1;   // 0..3 -> 32 rows each
  const int waveN = wave & 1;    // 0..1 -> 64 cols each
  const int m16   = lane & 15;
  const int hi    = lane >> 4;

  v8f acc[2][4];
  v8f zero = {};
#pragma unroll
  for (int i = 0; i < 2; ++i)
#pragma unroll
    for (int j = 0; j < 4; ++j) acc[i][j] = zero;

  // cooperative staging: each thread owns 32B of the A tile and 32B of the B tile
  const int iRow = tid >> 1;   // 0..127
  const int part = tid & 1;    // halves of the 32-element k-window
  const u16* gA0 = A  + (size_t)(rowBase + iRow) * K + part * 16;
  const u16* gB0 = Bt + (size_t)(colBase + iRow) * K + part * 16;
  const u32 laA[2] = { lds_addr(&lA[0][iRow * LDSH + part * 16]),
                       lds_addr(&lA[1][iRow * LDSH + part * 16]) };
  const u32 laB[2] = { lds_addr(&lB[0][iRow * LDSH + part * 16]),
                       lds_addr(&lB[1][iRow * LDSH + part * 16]) };

  auto issueTile = [&](int kb, int buf) {
    async_b128(laA[buf],      gA0 + kb);
    async_b128(laA[buf] + 16, gA0 + kb + 8);
    async_b128(laB[buf],      gB0 + kb);
    async_b128(laB[buf] + 16, gB0 + kb + 8);
  };

  issueTile(0, 0);
  const int NT = K >> 5;
  for (int ki = 0; ki < NT; ++ki) {
    const int cur = ki & 1;
    const bool pf = (ki + 1 < NT);
    if (pf) {
      issueTile((ki + 1) << 5, cur ^ 1);   // prefetch next tile into other buffer
      WAIT_ASYNC(4);                       // 8 in flight -> current tile's 4 done
    } else {
      WAIT_ASYNC(0);
    }
    __syncthreads();                       // all waves' copies for `cur` complete

    union Frag { v16bf v; uint4 q[2]; };
    Frag aF[2], bF[4];
    const u16* bufA = lA[cur];
    const u16* bufB = lB[cur];
#pragma unroll
    for (int mi = 0; mi < 2; ++mi) {
      int m = waveM * 32 + mi * 16 + m16;
      aF[mi].q[0] = *(const uint4*)&bufA[m * LDSH + 16 * hi];       // K 16*hi..+7
      aF[mi].q[1] = *(const uint4*)&bufA[m * LDSH + 16 * hi + 8];   // K 16*hi+8..+15
    }
#pragma unroll
    for (int ni = 0; ni < 4; ++ni) {
      int n = waveN * 64 + ni * 16 + m16;
      bF[ni].q[0] = *(const uint4*)&bufB[n * LDSH + 16 * hi];       // K 16*hi..+7
      bF[ni].q[1] = *(const uint4*)&bufB[n * LDSH + 16 * hi + 8];   // K 16*hi+8..+15
    }
#pragma unroll
    for (int mi = 0; mi < 2; ++mi)
#pragma unroll
      for (int ni = 0; ni < 4; ++ni)
        acc[mi][ni] = __builtin_amdgcn_wmma_f32_16x16x32_bf16(
            false, aF[mi].v, false, bF[ni].v, (short)0, acc[mi][ni], false, false);
    __syncthreads();   // reads of buf `cur` done before it is overwritten next+1 iter
  }

  // epilogue: D layout — element r of lane = D[mBase + 8*hi + r][nBase + m16]
#pragma unroll
  for (int mi = 0; mi < 2; ++mi) {
#pragma unroll
    for (int ni = 0; ni < 4; ++ni) {
      int col   = colBase + waveN * 64 + ni * 16 + m16;
      int rbase = rowBase + waveM * 32 + mi * 16 + 8 * hi;
#pragma unroll
      for (int r = 0; r < 8; ++r) {
        int row = rbase + r;
        float v = acc[mi][ni][r];
        size_t idx = (size_t)row * N + col;
        if (EPI == 0) {
          C[idx] = v;
        } else if (EPI == 1) {
          C[idx] = v + bias[col];
        } else {
          float rr = v + resid[idx];
          C[idx] = rr;
          outh[idx] = f32_bf16(rr);
        }
      }
    }
  }
}

// ---------------- dt = softplus(dt_raw + dt_bias) ----------------
__global__ void dt_kernel(const float* __restrict__ zx, const float* __restrict__ dt_bias,
                          float* __restrict__ dtb) {
  int idx = blockIdx.x * 256 + threadIdx.x;     // T_TOK * NHEADS
  int t = idx >> 5, hh = idx & 31;
  float v = zx[(size_t)t * NP_IN + (D_INNER + CONV_DIM) + hh] + dt_bias[hh];
  dtb[idx] = (v > 20.f) ? v : log1pf(__expf(v));
}

// ---------------- depthwise causal conv (width 4) + SiLU, split xs / BC ----------------
__global__ __launch_bounds__(256) void conv_kernel(
    const float* __restrict__ zx, const float* __restrict__ conv_w,
    const float* __restrict__ conv_b, float* __restrict__ xs, float* __restrict__ BC) {
  const int c = blockIdx.x * 256 + threadIdx.x;  // < CONV_DIM
  const int b = blockIdx.y;
  const float w0 = conv_w[c * 4 + 0], w1 = conv_w[c * 4 + 1];
  const float w2 = conv_w[c * 4 + 2], w3 = conv_w[c * 4 + 3];
  const float cb = conv_b[c];
  float x0 = 0.f, x1 = 0.f, x2 = 0.f;
  const float* src = zx + (size_t)b * SEQLEN * NP_IN + D_INNER + c;
  for (int l = 0; l < SEQLEN; ++l) {
    float xin = src[(size_t)l * NP_IN];
    float a = x0 * w0 + x1 * w1 + x2 * w2 + xin * w3 + cb;
    float s = siluf_(a);
    size_t t = (size_t)b * SEQLEN + l;
    if (c < D_INNER) xs[t * D_INNER + c] = s;
    else             BC[t * 256 + (c - D_INNER)] = s;
    x0 = x1; x1 = x2; x2 = xin;
  }
}

// ---------------- selective scan: one workgroup per (batch, head) ----------------
// 256 threads: p = tid>>2 (0..63 head dim), q = tid&3 (32-wide D_STATE slice).
// B/C for step l+1 stream into the other LDS buffer via async-to-LDS while step l computes.
__global__ __launch_bounds__(256) void scan_kernel(
    const float* __restrict__ xs, const float* __restrict__ BC, const float* __restrict__ dt,
    const float* __restrict__ A_log, const float* __restrict__ D_skip, float* __restrict__ y) {
  __shared__ float s_bc[2][256];   // [0..127]=B, [128..255]=C
  const int wg = blockIdx.x;
  const int b = wg >> 5, hh = wg & 31;
  const int tid = threadIdx.x;
  const int q = tid & 3, p = tid >> 2;
  const float Ah  = -__expf(A_log[hh]);
  const float Dsk = D_skip[hh];

  float h[32];
#pragma unroll
  for (int j = 0; j < 32; ++j) h[j] = 0.f;

  const size_t t0 = (size_t)b * SEQLEN;
  const u32 lbc[2] = { lds_addr(&s_bc[0][tid]), lds_addr(&s_bc[1][tid]) };

  async_b32(lbc[0], BC + t0 * 256 + tid);          // stage step 0
  float x_r  = xs[t0 * D_INNER + hh * HEADDIM + p];
  float dt_r = dt[t0 * NHEADS + hh];

  for (int l = 0; l < SEQLEN; ++l) {
    const int cur = l & 1;
    const size_t t = t0 + l;
    const bool pf = (l + 1 < SEQLEN);
    if (pf) async_b32(lbc[cur ^ 1], BC + (t + 1) * 256 + tid);   // prefetch step l+1

    const size_t tn = pf ? (t + 1) : t;
    float x_n  = xs[tn * D_INNER + hh * HEADDIM + p];
    float dt_n = dt[tn * NHEADS + hh];

    if (pf) WAIT_ASYNC(1); else WAIT_ASYNC(0);     // step l's copy complete
    __syncthreads();

    const float dA   = __expf(dt_r * Ah);
    const float coef = dt_r * x_r;
    float accv = 0.f;
    const float4* B4 = (const float4*)(&s_bc[cur][q * 32]);
    const float4* C4 = (const float4*)(&s_bc[cur][128 + q * 32]);
#pragma unroll
    for (int j4 = 0; j4 < 8; ++j4) {
      float4 bv = B4[j4], cv = C4[j4];
      h[4 * j4 + 0] = fmaf(h[4 * j4 + 0], dA, coef * bv.x); accv = fmaf(h[4 * j4 + 0], cv.x, accv);
      h[4 * j4 + 1] = fmaf(h[4 * j4 + 1], dA, coef * bv.y); accv = fmaf(h[4 * j4 + 1], cv.y, accv);
      h[4 * j4 + 2] = fmaf(h[4 * j4 + 2], dA, coef * bv.z); accv = fmaf(h[4 * j4 + 2], cv.z, accv);
      h[4 * j4 + 3] = fmaf(h[4 * j4 + 3], dA, coef * bv.w); accv = fmaf(h[4 * j4 + 3], cv.w, accv);
    }
    // reduce the 4 D_STATE slices (adjacent lanes, wave32)
    accv += __shfl_xor(accv, 1);
    accv += __shfl_xor(accv, 2);
    if (q == 0) y[t * D_INNER + hh * HEADDIM + p] = accv + Dsk * x_r;
    __syncthreads();   // buf `cur` free for the write issued at step l+1
    x_r = x_n; dt_r = dt_n;
  }
}

// ---------------- y * silu(z), RMSNorm, emit bf16 for out_proj ----------------
__global__ __launch_bounds__(256) void gate_rms_kernel(
    const float* __restrict__ y, const float* __restrict__ zx,
    const float* __restrict__ rms_w, u16* __restrict__ ynbf) {
  __shared__ float red[256];
  const int t = blockIdx.x, tid = threadIdx.x;
  float g[8];
  float ss = 0.f;
#pragma unroll
  for (int i = 0; i < 8; ++i) {
    int c = i * 256 + tid;
    float yv = y[(size_t)t * D_INNER + c];
    float zv = zx[(size_t)t * NP_IN + c];
    g[i] = yv * siluf_(zv);
    ss += g[i] * g[i];
  }
  red[tid] = ss;
  __syncthreads();
  for (int o = 128; o > 0; o >>= 1) { if (tid < o) red[tid] += red[tid + o]; __syncthreads(); }
  float r = rsqrtf(red[0] / (float)D_INNER + EPS);
#pragma unroll
  for (int i = 0; i < 8; ++i) {
    int c = i * 256 + tid;
    ynbf[(size_t)t * D_INNER + c] = f32_bf16(g[i] * r * rms_w[c]);
  }
}

// ---------------- GLU gate + LayerNorm -> output ----------------
__global__ __launch_bounds__(256) void final_kernel(
    const float* __restrict__ R, const float* __restrict__ G,
    const float* __restrict__ ln_w, const float* __restrict__ ln_b, float* __restrict__ out) {
  __shared__ float red[256];
  const int t = blockIdx.x, tid = threadIdx.x;
  float yv[4];
  float s = 0.f;
#pragma unroll
  for (int i = 0; i < 4; ++i) {
    int c = i * 256 + tid;
    float rv = R[(size_t)t * D_MODEL + c];
    float a  = G[(size_t)t * (2 * D_MODEL) + c];
    float gt = G[(size_t)t * (2 * D_MODEL) + D_MODEL + c];
    yv[i] = rv + a * sigmoidf_(gt);
    s += yv[i];
  }
  red[tid] = s;
  __syncthreads();
  for (int o = 128; o > 0; o >>= 1) { if (tid < o) red[tid] += red[tid + o]; __syncthreads(); }
  float mu = red[0] / (float)D_MODEL;
  __syncthreads();
  float s2 = 0.f;
#pragma unroll
  for (int i = 0; i < 4; ++i) { float d = yv[i] - mu; s2 += d * d; }
  red[tid] = s2;
  __syncthreads();
  for (int o = 128; o > 0; o >>= 1) { if (tid < o) red[tid] += red[tid + o]; __syncthreads(); }
  float inv = rsqrtf(red[0] / (float)D_MODEL + EPS);
#pragma unroll
  for (int i = 0; i < 4; ++i) {
    int c = i * 256 + tid;
    out[(size_t)t * D_MODEL + c] = (yv[i] - mu) * inv * ln_w[c] + ln_b[c];
  }
}

// ---------------- host launcher ----------------
extern "C" void kernel_launch(void* const* d_in, const int* in_sizes, int n_in,
                              void* d_out, int out_size, void* d_ws, size_t ws_size,
                              hipStream_t stream) {
  const float* x       = (const float*)d_in[0];   // T x 1024
  const float* W_in    = (const float*)d_in[1];   // 1024 x 4384
  const float* conv_w  = (const float*)d_in[2];   // 2304 x 4
  const float* conv_b  = (const float*)d_in[3];
  const float* dt_bias = (const float*)d_in[4];
  const float* A_log   = (const float*)d_in[5];
  const float* D_skip  = (const float*)d_in[6];
  const float* rms_w   = (const float*)d_in[7];
  const float* W_out   = (const float*)d_in[8];   // 2048 x 1024
  const float* glu_w   = (const float*)d_in[9];   // 1024 x 2048
  const float* glu_b   = (const float*)d_in[10];
  const float* ln_w    = (const float*)d_in[11];
  const float* ln_b    = (const float*)d_in[12];
  float* out = (float*)d_out;

  char* w = (char*)d_ws;
  auto carve = [&](size_t bytes) { char* p = w; w += (bytes + 255) & ~(size_t)255; return p; };

  u16*   Xbf   = (u16*)  carve((size_t)T_TOK * D_MODEL * 2);
  u16*   WtIn  = (u16*)  carve((size_t)NP_IN * D_MODEL * 2);
  u16*   WtOut = (u16*)  carve((size_t)D_MODEL * D_INNER * 2);
  u16*   WtGlu = (u16*)  carve((size_t)(2 * D_MODEL) * D_MODEL * 2);
  float* ZX    = (float*)carve((size_t)T_TOK * NP_IN * 4);
  float* DT    = (float*)carve((size_t)T_TOK * NHEADS * 4);
  float* XS    = (float*)carve((size_t)T_TOK * D_INNER * 4);
  float* BC    = (float*)carve((size_t)T_TOK * 256 * 4);
  float* Y     = (float*)carve((size_t)T_TOK * D_INNER * 4);
  u16*   YNbf  = (u16*)  carve((size_t)T_TOK * D_INNER * 2);
  float* R     = (float*)carve((size_t)T_TOK * D_MODEL * 4);
  u16*   Rbf   = (u16*)  carve((size_t)T_TOK * D_MODEL * 2);
  float* G     = (float*)carve((size_t)T_TOK * (2 * D_MODEL) * 4);
  (void)ws_size; (void)in_sizes; (void)n_in; (void)out_size;

  // 1) convert / transpose weights & activations to bf16 layouts
  cvt_bf16_kernel<<<(T_TOK * D_MODEL) / 256, 256, 0, stream>>>(x, Xbf, (size_t)T_TOK * D_MODEL);
  transpose_bf16_kernel<<<dim3(NP_IN / 32, D_MODEL / 32), 256, 0, stream>>>(W_in, WtIn, D_MODEL, D_IN_PROJ);
  transpose_bf16_kernel<<<dim3(D_MODEL / 32, D_INNER / 32), 256, 0, stream>>>(W_out, WtOut, D_INNER, D_MODEL);
  transpose_bf16_kernel<<<dim3((2 * D_MODEL) / 32, D_MODEL / 32), 256, 0, stream>>>(glu_w, WtGlu, D_MODEL, 2 * D_MODEL);

  // 2) in_proj GEMM (WMMA): ZX[T x 4480] = Xbf @ WtIn^T
  gemm_bf16_wmma<0><<<dim3(NP_IN / 128, T_TOK / 128), 256, 0, stream>>>(
      Xbf, WtIn, ZX, NP_IN, D_MODEL, nullptr, nullptr, nullptr);

  // 3) dt softplus, depthwise conv + SiLU
  dt_kernel<<<(T_TOK * NHEADS) / 256, 256, 0, stream>>>(ZX, dt_bias, DT);
  conv_kernel<<<dim3(CONV_DIM / 256, BATCH), 256, 0, stream>>>(ZX, conv_w, conv_b, XS, BC);

  // 4) selective scan
  scan_kernel<<<BATCH * NHEADS, 256, 0, stream>>>(XS, BC, DT, A_log, D_skip, Y);

  // 5) gate + RMSNorm -> bf16
  gate_rms_kernel<<<T_TOK, 256, 0, stream>>>(Y, ZX, rms_w, YNbf);

  // 6) out_proj GEMM + residual (fp32 R and bf16 Rbf)
  gemm_bf16_wmma<2><<<dim3(D_MODEL / 128, T_TOK / 128), 256, 0, stream>>>(
      YNbf, WtOut, R, D_MODEL, D_INNER, nullptr, x, Rbf);

  // 7) GLU GEMM + bias
  gemm_bf16_wmma<1><<<dim3((2 * D_MODEL) / 128, T_TOK / 128), 256, 0, stream>>>(
      Rbf, WtGlu, G, 2 * D_MODEL, D_MODEL, glu_b, nullptr, nullptr);

  // 8) GLU gate + LayerNorm -> output
  final_kernel<<<T_TOK, 256, 0, stream>>>(R, G, ln_w, ln_b, out);
}
